// PrismModel_44324062494750
// MI455X (gfx1250) — matrix operand order repaired and checked by reading, not compiled, not run on hardware
//
#include <hip/hip_runtime.h>
#include <stdint.h>

#define LDS3 __attribute__((address_space(3)))

typedef __bf16    bf16x16 __attribute__((ext_vector_type(16)));
typedef __bf16    bf16x2  __attribute__((ext_vector_type(2)));
typedef float     f32x8   __attribute__((ext_vector_type(8)));
typedef float     f32x4   __attribute__((ext_vector_type(4)));
typedef unsigned  u32x4   __attribute__((ext_vector_type(4)));

#define KC   32      // K-chunk (matches WMMA K for bf16)
#define KCP  40      // padded K pitch of weight stage (halves, mult of 8 for b128)
#define HP   264     // padded pitch of 256-wide activation tiles (halves)
#define FP   520     // padded pitch of 512-wide feat tile (halves)
#define B_   128
#define H_   256

// A fragment: 16x32 bf16, M rows [mBase,mBase+16), K cols [kBase,kBase+32).
// ISA layout: lanes 0-15 -> M=lane, K groups {0-7,16-23}; lanes 16-31 -> +8.
__device__ __forceinline__ bf16x16 load_a(const LDS3 __bf16* h, int pitch,
                                          int mBase, int kBase) {
  int lane = threadIdx.x & 31;
  int m  = mBase + (lane & 15);
  int kb = kBase + ((lane >> 4) << 3);
  union { u32x4 u[2]; bf16x16 v; } f;
  f.u[0] = *(const LDS3 u32x4*)(h + m * pitch + kb);        // K kb..kb+7
  f.u[1] = *(const LDS3 u32x4*)(h + m * pitch + kb + 16);   // K kb+16..kb+23
  return f.v;
}

// B fragment: 32x16 bf16 from K-transposed stage [n][k] (pitch KCP halves).
// lanes 0-15 -> K groups {0-7,16-23} of column n; lanes 16-31 -> +8.
__device__ __forceinline__ bf16x16 load_b(const LDS3 __bf16* w, int nBase) {
  int lane = threadIdx.x & 31;
  int n  = nBase + (lane & 15);
  int kb = (lane >> 4) << 3;
  union { u32x4 u[2]; bf16x16 v; } f;
  f.u[0] = *(const LDS3 u32x4*)(w + n * KCP + kb);
  f.u[1] = *(const LDS3 u32x4*)(w + n * KCP + kb + 16);
  return f.v;
}

// Cooperative fetch of a 32x256 fp32 chunk. Thread owns 2 consecutive K rows
// x 16 columns (4 col-groups of 4). Lanes 0-15 cover 256B contiguous.
__device__ __forceinline__ void fetch_chunk(const float* __restrict__ W, int kc,
                                            f32x4* r) {
  int t = threadIdx.x;
  int kk = (t >> 4) << 1;
  int colBase = (t & 15) << 2;
  const float* s0 = W + (size_t)(kc * KC + kk) * H_ + colBase;
  const float* s1 = s0 + H_;
  #pragma unroll
  for (int i = 0; i < 4; ++i) {
    r[i]     = *(const f32x4*)(s0 + (i << 6));
    r[i + 4] = *(const f32x4*)(s1 + (i << 6));
  }
}

// Convert fp32 -> bf16 (paired, enables v_cvt_pk_bf16_f32) and store
// K-transposed: two consecutive K per 32-bit ds_store.
__device__ __forceinline__ void store_chunk(LDS3 __bf16* stg, const f32x4* r) {
  int t = threadIdx.x;
  int kk = (t >> 4) << 1;
  int colBase = (t & 15) << 2;
  #pragma unroll
  for (int i = 0; i < 4; ++i) {
    #pragma unroll
    for (int j = 0; j < 4; ++j) {
      int n = colBase + (i << 6) + j;
      bf16x2 p;
      p.x = (__bf16)r[i][j];
      p.y = (__bf16)r[i + 4][j];
      *(LDS3 bf16x2*)(stg + n * KCP + kk) = p;
    }
  }
}

// hout[128x256] = relu(hin[128xK] @ W[Kx256] + bias), bf16 activations in LDS.
// 8 waves; wave w owns M rows [16w,16w+16), 16 N-tiles of WMMA accumulators.
// All 32 B-fragment ds_load_b128s issue before the WMMA chain; a sched_barrier
// pins them there so the fragments stay live in distinct registers and the
// WMMAs drain with partial s_wait_dscnt instead of full LDS round-trips.
__device__ __forceinline__ void gemm_relu(
    const float* __restrict__ W, const float* __restrict__ bias,
    const LDS3 __bf16* hin, int pitchIn,
    LDS3 __bf16* hout, LDS3 __bf16* wstage, int K) {
  const int t    = threadIdx.x;
  const int lane = t & 31;
  const int wv   = t >> 5;
  const int nk   = K / KC;

  f32x8 acc[16];
  f32x8 z = {0.f, 0.f, 0.f, 0.f, 0.f, 0.f, 0.f, 0.f};
  #pragma unroll
  for (int nt = 0; nt < 16; ++nt) acc[nt] = z;

  f32x4 regs[8];
  fetch_chunk(W, 0, regs);
  store_chunk(wstage, regs);
  __syncthreads();

  #pragma unroll 1
  for (int kc = 0; kc < nk; ++kc) {
    if (kc + 1 < nk) fetch_chunk(W, kc + 1, regs);
    if (kc + 2 < nk)
      __builtin_prefetch(W + (size_t)((kc + 2) * KC + ((t >> 4) << 1)) * H_ + ((t & 15) << 2), 0, 1);

    const LDS3 __bf16* wcur = wstage + (kc & 1) * (H_ * KCP);
    bf16x16 a = load_a(hin, pitchIn, wv << 4, kc * KC);

    bf16x16 bfr[16];
    #pragma unroll
    for (int nt = 0; nt < 16; ++nt) bfr[nt] = load_b(wcur, nt << 4);

    // Pin all B loads above, all WMMAs below.
    __builtin_amdgcn_sched_barrier(0);

    #pragma unroll
    for (int nt = 0; nt < 16; ++nt)
      acc[nt] = __builtin_amdgcn_wmma_f32_16x16x32_bf16(
          false, a, false, bfr[nt], (short)0, acc[nt], false, false);

    if (kc + 1 < nk) store_chunk(wstage + ((kc + 1) & 1) * (H_ * KCP), regs);
    __syncthreads();
  }

  // Epilogue: C layout -> lane gives N=lane&15, rows 8*(lane>>4)+r.
  int mBase = (wv << 4) + ((lane >> 4) << 3);
  int nOff  = lane & 15;
  #pragma unroll
  for (int nt = 0; nt < 16; ++nt) {
    int n = (nt << 4) + nOff;
    float bn = bias[n];
    #pragma unroll
    for (int r = 0; r < 8; ++r) {
      float v = acc[nt][r] + bn;
      v = v > 0.f ? v : 0.f;
      hout[(mBase + r) * HP + n] = (__bf16)v;
    }
  }
}

// ---------------- Kernel 1: att / rad deep MLPs ----------------
__global__ __launch_bounds__(256)
void deep_mlp_kernel(const float* __restrict__ pos, const float* __restrict__ ue,
                     const float* __restrict__ bs,  const float* __restrict__ add,
                     const float* __restrict__ attW1, const float* __restrict__ attB1,
                     const float* __restrict__ attWh, const float* __restrict__ attBh,
                     const float* __restrict__ radW1, const float* __restrict__ radB1,
                     const float* __restrict__ radWh, const float* __restrict__ radBh,
                     __bf16* __restrict__ featG) {
  __shared__ __align__(16) float  xs[B_ * 20];
  __shared__ __align__(16) __bf16 hA[B_ * HP];
  __shared__ __align__(16) __bf16 hB[B_ * HP];
  __shared__ __align__(16) __bf16 wstage[2 * H_ * KCP];

  LDS3 __bf16* hAs = (LDS3 __bf16*)hA;
  LDS3 __bf16* hBs = (LDS3 __bf16*)hB;
  LDS3 __bf16* wst = (LDS3 __bf16*)wstage;

  const int t   = threadIdx.x;
  const int net = blockIdx.x;      // 0 = attenuation, 1 = radiance
  const float* W1 = net ? radW1 : attW1;
  const float* B1 = net ? radB1 : attB1;
  const float* Wh = net ? radWh : attWh;
  const float* Bh = net ? radBh : attBh;

  // Build x = concat(pos, ue, bs, add) : [128,19]
  for (int idx = t; idx < B_ * 19; idx += 256) {
    int b = idx / 19, f = idx % 19;
    float v;
    if (f < 3)      v = pos[b * 3 + f];
    else if (f < 5) v = ue[b * 2 + f - 3];
    else if (f < 9) v = bs[b * 4 + f - 5];
    else            v = add[b * 10 + f - 9];
    xs[b * 20 + f] = v;
  }
  __syncthreads();

  // Layer 1 (K=19): scalar, negligible FLOPs
  for (int o = t; o < B_ * H_; o += 256) {
    int b = o >> 8, n = o & 255;
    float acc = B1[n];
    #pragma unroll
    for (int k = 0; k < 19; ++k) acc += xs[b * 20 + k] * W1[k * H_ + n];
    acc = acc > 0.f ? acc : 0.f;
    hAs[b * HP + n] = (__bf16)acc;
  }
  __syncthreads();

  LDS3 __bf16* cur = hAs;
  LDS3 __bf16* nxt = hBs;
  for (int l = 0; l < 7; ++l) {
    gemm_relu(Wh + (size_t)l * H_ * H_, Bh + (size_t)l * H_, cur, HP, nxt, wst, H_);
    __syncthreads();
    LDS3 __bf16* tmp = cur; cur = nxt; nxt = tmp;
  }

  // Write half of feat[128,512] (bf16) to workspace
  for (int idx = t; idx < B_ * H_; idx += 256) {
    int b = idx >> 8, n = idx & 255;
    featG[b * 512 + net * H_ + n] = cur[b * HP + n];
  }
}

// ---------------- Kernel 2: fused per-subcarrier prism MLP ----------------
__global__ __launch_bounds__(256)
void prism_kernel(const __bf16* __restrict__ featG,
                  const float* __restrict__ W1, const float* __restrict__ B1,
                  const float* __restrict__ W2, const float* __restrict__ B2,
                  const float* __restrict__ W3, const float* __restrict__ B3,
                  float* __restrict__ sub) {
  __shared__ __align__(16) __bf16 featL[B_ * FP];       // 133 KB, reused as h2
  __shared__ __align__(16) __bf16 h1s[B_ * HP];         //  66 KB
  __shared__ __align__(16) __bf16 wstage[2 * H_ * KCP]; //  40 KB double buffer
  __shared__ __align__(16) float  w3s[H_];

  LDS3 __bf16* featLs = (LDS3 __bf16*)featL;
  LDS3 __bf16* h1p    = (LDS3 __bf16*)h1s;
  LDS3 __bf16* wst    = (LDS3 __bf16*)wstage;

  const int t = threadIdx.x;
  const int c = blockIdx.x;

  // Stage feat [128,512] bf16 into LDS (padded pitch)
  for (int idx = t; idx < B_ * 512 / 8; idx += 256) {
    int b = idx >> 6, g = (idx & 63) << 3;
    *(LDS3 u32x4*)(featLs + b * FP + g) = *(const u32x4*)(featG + b * 512 + g);
  }
  __syncthreads();

  // h1 = relu(feat @ W1[c] + b1[c])   (K=512)
  gemm_relu(W1 + (size_t)c * 512 * H_, B1 + (size_t)c * H_, featLs, FP, h1p, wst, 512);
  __syncthreads();

  // h2 = relu(h1 @ W2[c] + b2[c])     (K=256), reuse featL storage for h2
  LDS3 __bf16* h2p = featLs;
  gemm_relu(W2 + (size_t)c * H_ * H_, B2 + (size_t)c * H_, h1p, HP, h2p, wst, H_);

  if (t < H_) w3s[t] = W3[(size_t)c * H_ + t];
  __syncthreads();

  // sub[:,c] = h2 @ w3[c] + b3[c]
  if (t < B_) {
    float acc = 0.f;
    #pragma unroll 4
    for (int k = 0; k < H_; ++k) acc += (float)h2p[t * HP + k] * w3s[k];
    sub[(size_t)t * 1024 + c] = acc + B3[c];
  }
}

// ---------------- Kernel 3: mimo head ----------------
__global__ __launch_bounds__(128)
void mimo_kernel(const float* __restrict__ sub, const float* __restrict__ mW,
                 const float* __restrict__ mB, float* __restrict__ out) {
  int b = threadIdx.x;
  if (b < B_) {
    float s = 0.f;
    for (int c = 0; c < 1024; ++c) s += sub[(size_t)b * 1024 + c];
    float m = s * (1.0f / 1024.0f);
    #pragma unroll
    for (int j = 0; j < 8; ++j) out[b * 8 + j] = m * mW[j] + mB[j];
  }
}

extern "C" void kernel_launch(void* const* d_in, const int* in_sizes, int n_in,
                              void* d_out, int out_size, void* d_ws, size_t ws_size,
                              hipStream_t stream) {
  const float* pos   = (const float*)d_in[0];
  const float* ue    = (const float*)d_in[1];
  const float* bs    = (const float*)d_in[2];
  const float* add   = (const float*)d_in[3];
  const float* attW1 = (const float*)d_in[4];
  const float* attB1 = (const float*)d_in[5];
  const float* attWh = (const float*)d_in[6];
  const float* attBh = (const float*)d_in[7];
  const float* radW1 = (const float*)d_in[8];
  const float* radB1 = (const float*)d_in[9];
  const float* radWh = (const float*)d_in[10];
  const float* radBh = (const float*)d_in[11];
  const float* pW1   = (const float*)d_in[12];
  const float* pB1   = (const float*)d_in[13];
  const float* pW2   = (const float*)d_in[14];
  const float* pB2   = (const float*)d_in[15];
  const float* pW3   = (const float*)d_in[16];
  const float* pB3   = (const float*)d_in[17];
  const float* mW    = (const float*)d_in[18];
  const float* mB    = (const float*)d_in[19];

  __bf16* featG = (__bf16*)d_ws;               // 128*512 bf16 = 128 KB
  float* sub  = (float*)d_out;                  // [128,1024]
  float* mimo = sub + 128 * 1024;               // [128,8]

  deep_mlp_kernel<<<2, 256, 0, stream>>>(pos, ue, bs, add,
                                         attW1, attB1, attWh, attBh,
                                         radW1, radB1, radWh, radBh, featG);
  prism_kernel<<<1024, 256, 0, stream>>>(featG, pW1, pB1, pW2, pB2, pW3, pB3, sub);
  mimo_kernel<<<1, 128, 0, stream>>>(sub, mW, mB, mimo);
}
